// HybridLSTMCell_730144440614
// MI455X (gfx1250) — compile-verified, matrix operand order checked by reading, not run on hardware
//
#include <hip/hip_runtime.h>
#include <hip/hip_bf16.h>

// ---------------------------------------------------------------------------
// HybridLSTMCell on MI455X (gfx1250, wave32, WMMA, async Global->LDS DMA)
//
//   x = round(2048*[inp,h]) is loop-invariant => the 32-step ternary spike
//   stream lin_t is precomputed. Heavy work: 4 gates x 32 steps of
//   [256x2048]x[2048x1024] GEMMs (~137 GFLOP) via v_wmma_f32_16x16x32_f16.
//
// Roofline: weights (16MB f16) + spikes (32MB f16) are L2-resident (192MB).
//   T'=4 temporal blocking reuses each weight B-fragment across 4 timestep
//   accumulators (weight L2 traffic /4, ~2.1 GB) while the per-(b,h)
//   IF-neuron state stays register-resident. A tiles staged in LDS in
//   K-panels of 512 (4 x 16 x 512 f16 = 64 KB), double-buffered and filled
//   with global_load_async_to_lds_b128 (ASYNCcnt) so the DMA of panel q+1
//   overlaps the 256 WMMAs of panel q — no VGPR staging, no copy bubble.
//
// Workspace: 48 MB  (lin: 32 MB @ offset 0, f16 weights: 16 MB @ 32 MB)
// ---------------------------------------------------------------------------

typedef __attribute__((ext_vector_type(16))) _Float16 v16h;
typedef __attribute__((ext_vector_type(8)))  _Float16 v8h;
typedef __attribute__((ext_vector_type(8)))  float    v8f;

constexpr int BB = 256;     // batch
constexpr int II = 1024;    // input dim
constexpr int HH = 1024;    // hidden dim
constexpr int DD = II + HH; // 2048 (K)
constexpr int TT = 32;      // timesteps
constexpr int TP = 4;       // timesteps per K-pass (temporal blocking)
constexpr int KP = 512;     // K-panel staged in LDS
constexpr int NPASS = (TT / TP) * (DD / KP);  // 32 pipeline passes

constexpr float VTH_SG = 2048.0f;
constexpr float VTH_T1 = 1024.0f;
constexpr float VTH_T2 = 512.0f;

#ifndef __has_builtin
#define __has_builtin(x) 0
#endif

// Wait for all of this wave's async Global<->LDS ops (ASYNCcnt == 0).
__device__ inline void wait_async0() {
#if __has_builtin(__builtin_amdgcn_s_wait_asynccnt)
  __builtin_amdgcn_s_wait_asynccnt(0);
#else
  asm volatile("s_wait_asynccnt 0x0" ::: "memory");
#endif
}

// One 16-byte async Global->LDS transfer (per active lane). VDST carries the
// wave-relative LDS byte address (= low 32 bits of the generic pointer, since
// the LDS aperture encodes LDS_ADDR in ADDR[31:0]).
__device__ inline void async_g2l_b128(uint32_t lds_byte, const void* gaddr) {
  asm volatile("global_load_async_to_lds_b128 %0, %1, off"
               :
               : "v"(lds_byte), "v"((unsigned long long)(uintptr_t)gaddr)
               : "memory");
}

// ---------------------------------------------------------------------------
// Kernel 1: precompute lin[t][b][d] = spike * sign(x), x = round(2048*[inp,h])
// ---------------------------------------------------------------------------
__global__ __launch_bounds__(256) void spikegen_kernel(
    const float* __restrict__ inp, const float* __restrict__ hbuf,
    _Float16* __restrict__ lin)
{
  const int idx = blockIdx.x * 256 + threadIdx.x;     // over BB*DD
  const int b = idx / DD;
  const int d = idx - b * DD;
  const float v = (d < II) ? inp[(size_t)b * II + d]
                           : hbuf[(size_t)b * HH + (d - II)];
  const float x  = rintf(2048.0f * v);                // RNE, matches jnp.round
  const float ax = fabsf(x);
  const float sg = (x > 0.0f ? 1.0f : 0.0f) - (x < 0.0f ? 1.0f : 0.0f);

  float vmem = 1.0f;                                  // carry0 = ones
  #pragma unroll
  for (int t = 0; t < TT; ++t) {
    vmem += ax;
    const float spk = (vmem > VTH_SG) ? 1.0f : 0.0f;
    vmem -= VTH_SG * spk;
    lin[((size_t)t * BB + b) * (size_t)DD + d] = (_Float16)(spk * sg);
  }
}

// ---------------------------------------------------------------------------
// Kernel 2: convert the 4 gate weight matrices to f16, layout [g][H][D].
// ---------------------------------------------------------------------------
__global__ __launch_bounds__(256) void wconvert_kernel(
    const float* __restrict__ Wi, const float* __restrict__ Wf,
    const float* __restrict__ Wc, const float* __restrict__ Wo,
    _Float16* __restrict__ w16)
{
  const size_t idx = (size_t)blockIdx.x * 256 + threadIdx.x; // over 4*H*D
  const size_t g = idx / ((size_t)HH * DD);
  const size_t r = idx - g * ((size_t)HH * DD);
  const float* src = (g == 0) ? Wi : (g == 1) ? Wf : (g == 2) ? Wc : Wo;
  w16[idx] = (_Float16)src[r];
}

// ---------------------------------------------------------------------------
// Fragment packing per ISA 7.12.2 (16-bit 16x32 A / 32x16 B, wave32):
//   lanes 0-15  : K = {k0+0..7,  k0+16..23}
//   lanes 16-31 : K = {k0+8..15, k0+24..31}
// ---------------------------------------------------------------------------
__device__ inline v16h ld_frag(const _Float16* __restrict__ p, int kbase)
{
  v8h lo = *(const v8h*)(p + kbase);
  v8h hi = *(const v8h*)(p + kbase + 16);
  v16h r;
  #pragma unroll
  for (int i = 0; i < 8; ++i) { r[i] = lo[i]; r[i + 8] = hi[i]; }
  return r;
}

#define WMMA_F16(acc, A, Bf)                                             \
  acc = __builtin_amdgcn_wmma_f32_16x16x32_f16(false, (A), false, (Bf),  \
                                               (short)0, (acc), false, false)

// One IF-neuron timestep on register state from the 4 gate sums.
__device__ inline void if_step(
    const v8f& AI, const v8f& AF, const v8f& AC, const v8f& AO,
    float bi, float bf, float bc, float bo,
    v8f& ig, v8f& fg, v8f& og, v8f& cp, v8f& cn,
    v8f& si, v8f& sf, v8f& sc, v8f& so)
{
  #pragma unroll
  for (int r = 0; r < 8; ++r) {
    ig[r] += AI[r] + bi;
    float sp = (ig[r] > VTH_T1) ? 1.0f : 0.0f;
    ig[r] -= VTH_T1 * sp;  si[r] += sp;

    fg[r] += AF[r] + bf;
    sp = (fg[r] > VTH_T1) ? 1.0f : 0.0f;
    fg[r] -= VTH_T1 * sp;  sf[r] += sp;

    og[r] += AO[r] + bo;
    sp = (og[r] > VTH_T1) ? 1.0f : 0.0f;
    og[r] -= VTH_T1 * sp;  so[r] += sp;

    const float drive = AC[r] + bc;
    cp[r] += drive;
    cn[r] -= drive;
    const float spp = (cp[r] > VTH_T2) ? 1.0f : 0.0f;
    const float spn = (cn[r] > VTH_T2) ? 1.0f : 0.0f;
    cp[r] += VTH_T2 * (spn - spp);
    cn[r] += VTH_T2 * (spp - spn);
    sc[r] += spp - spn;
  }
}

// ---------------------------------------------------------------------------
// Kernel 3: fused 4-gate GEMM + IF-neuron temporal scan, T'=4 blocking,
// double-buffered async Global->LDS pipeline over 32 (group,panel) passes.
// grid = (H/64, B/16); block = 128 (4 waves); each wave owns a 16x16 output
// tile for all 4 gates; 16 WMMAs per 32-wide K-chunk (4 gates x 4 timesteps).
// ---------------------------------------------------------------------------
__global__ __launch_bounds__(128) void fused_gemm_scan_kernel(
    const _Float16* __restrict__ lin,   // [T][B][D]
    const _Float16* __restrict__ w16,   // [4][H][D]
    const float* __restrict__ c_old,    // [B][H]
    const float* __restrict__ b_i, const float* __restrict__ b_f,
    const float* __restrict__ b_c, const float* __restrict__ b_o,
    float* __restrict__ out)            // h_new [B][H] then c_new [B][H]
{
  __shared__ _Float16 ldsA[2][TP * 16 * KP];   // 2 x 64 KB

  const int tid  = threadIdx.x;
  const int wave = tid >> 5;
  const int lane = tid & 31;

  const int m0   = blockIdx.y * 16;             // batch tile
  const int n0   = blockIdx.x * 64 + wave * 16; // hidden tile (per wave)
  const int n    = n0 + (lane & 15);
  const int koff = (lane & 16) ? 8 : 0;         // K-phase per ISA layout
  const int arow = lane & 15;

  const _Float16* wi = w16 + ((size_t)0 * HH + n) * DD;
  const _Float16* wf = w16 + ((size_t)1 * HH + n) * DD;
  const _Float16* wc = w16 + ((size_t)2 * HH + n) * DD;
  const _Float16* wo = w16 + ((size_t)3 * HH + n) * DD;

  const float bi = b_i[n], bf = b_f[n], bc = b_c[n], bo = b_o[n];

  // Issue this thread's 32 async 16B transfers for pass q into buffer `buf`.
  auto issue_panel = [&](int q, int buf) {
    const int ts_base = (q >> 2) * TP;
    const int p = q & 3;
    #pragma unroll
    for (int i = 0; i < 32; ++i) {
      const int u = tid + i * 128;        // 0..4095 uint4s
      const int row_lin = u >> 6;         // 64 uint4 per row
      const int col = (u & 63) << 3;      // half-offset within panel
      const int ts  = row_lin >> 4;       // which timestep tile
      const int row = row_lin & 15;
      const _Float16* src = lin +
          (((size_t)(ts_base + ts) * BB + m0 + row) * DD + p * KP + col);
      const uint32_t dst = (uint32_t)(uintptr_t)
          &ldsA[buf][(size_t)((ts << 4) | row) * KP + col];
      async_g2l_b128(dst, src);
    }
  };

  // IF-neuron state (carry0 = ones) + spike-count accumulators.
  v8f ig, fg, og, cp, cn, si, sf, sc, so;
  #pragma unroll
  for (int r = 0; r < 8; ++r) {
    ig[r] = 1.0f; fg[r] = 1.0f; og[r] = 1.0f; cp[r] = 1.0f; cn[r] = 1.0f;
    si[r] = 0.0f; sf[r] = 0.0f; sc[r] = 0.0f; so[r] = 0.0f;
  }

  // 4 gates x 4 timesteps of gate-sum accumulators (persist across panels).
  v8f ai0 = {}, ai1 = {}, ai2 = {}, ai3 = {};
  v8f af0 = {}, af1 = {}, af2 = {}, af3 = {};
  v8f ac0 = {}, ac1 = {}, ac2 = {}, ac3 = {};
  v8f ao0 = {}, ao1 = {}, ao2 = {}, ao3 = {};

  issue_panel(0, 0);   // prologue DMA

  for (int q = 0; q < NPASS; ++q) {
    const int p   = q & 3;     // panel within the 4-timestep group
    const int buf = q & 1;

    if (p == 0) {              // new timestep group: reset accumulators
      ai0 = (v8f){}; ai1 = (v8f){}; ai2 = (v8f){}; ai3 = (v8f){};
      af0 = (v8f){}; af1 = (v8f){}; af2 = (v8f){}; af3 = (v8f){};
      ac0 = (v8f){}; ac1 = (v8f){}; ac2 = (v8f){}; ac3 = (v8f){};
      ao0 = (v8f){}; ao1 = (v8f){}; ao2 = (v8f){}; ao3 = (v8f){};
    }

    wait_async0();       // this wave's DMA for `buf` has landed in LDS
    __syncthreads();     // all waves' DMA landed; prior reads of buf^1 done

    if (q + 1 < NPASS) issue_panel(q + 1, buf ^ 1);   // overlap with compute

    const _Float16* base = &ldsA[buf][0];
    const _Float16* a0p = base + (size_t)(0 * 16 + arow) * KP;
    const _Float16* a1p = base + (size_t)(1 * 16 + arow) * KP;
    const _Float16* a2p = base + (size_t)(2 * 16 + arow) * KP;
    const _Float16* a3p = base + (size_t)(3 * 16 + arow) * KP;

    for (int kc = 0; kc < KP / 32; ++kc) {
      const int kl = kc * 32 + koff;        // within panel
      const int kg = p * KP + kl;           // global K
      const v16h a0 = ld_frag(a0p, kl);
      const v16h a1 = ld_frag(a1p, kl);
      const v16h a2 = ld_frag(a2p, kl);
      const v16h a3 = ld_frag(a3p, kl);
      {
        const v16h bfr = ld_frag(wi, kg);
        WMMA_F16(ai0, a0, bfr); WMMA_F16(ai1, a1, bfr);
        WMMA_F16(ai2, a2, bfr); WMMA_F16(ai3, a3, bfr);
      }
      {
        const v16h bfr = ld_frag(wf, kg);
        WMMA_F16(af0, a0, bfr); WMMA_F16(af1, a1, bfr);
        WMMA_F16(af2, a2, bfr); WMMA_F16(af3, a3, bfr);
      }
      {
        const v16h bfr = ld_frag(wc, kg);
        WMMA_F16(ac0, a0, bfr); WMMA_F16(ac1, a1, bfr);
        WMMA_F16(ac2, a2, bfr); WMMA_F16(ac3, a3, bfr);
      }
      {
        const v16h bfr = ld_frag(wo, kg);
        WMMA_F16(ao0, a0, bfr); WMMA_F16(ao1, a1, bfr);
        WMMA_F16(ao2, a2, bfr); WMMA_F16(ao3, a3, bfr);
      }
    }

    if (p == 3) {   // group complete: 4 sequential IF-neuron updates
      if_step(ai0, af0, ac0, ao0, bi, bf, bc, bo, ig, fg, og, cp, cn, si, sf, sc, so);
      if_step(ai1, af1, ac1, ao1, bi, bf, bc, bo, ig, fg, og, cp, cn, si, sf, sc, so);
      if_step(ai2, af2, ac2, ao2, bi, bf, bc, bo, ig, fg, og, cp, cn, si, sf, sc, so);
      if_step(ai3, af3, ac3, ao3, bi, bf, bc, bo, ig, fg, og, cp, cn, si, sf, sc, so);
    }
  }

  // Finalize. C-layout: lanes 0-15 hold M=r, lanes 16-31 hold M=r+8, N=lane&15.
  const float invT = 1.0f / (float)TT;
  const int mhi = (lane >> 4) << 3;
  #pragma unroll
  for (int r = 0; r < 8; ++r) {
    const int bidx = m0 + r + mhi;
    const float avg_i = si[r] * invT;
    const float avg_f = sf[r] * invT;
    const float avg_c = sc[r] * invT;
    const float avg_o = so[r] * invT;
    const float cO = c_old[(size_t)bidx * HH + n];
    const float cN = avg_f * cO + avg_i * avg_c;
    const float act = fminf(fmaxf(0.5f * cN, -1.0f), 1.0f);
    out[(size_t)bidx * HH + n] = avg_o * act;                       // h_new
    out[(size_t)BB * HH + (size_t)bidx * HH + n] = cN;              // c_new
  }
}

// ---------------------------------------------------------------------------
extern "C" void kernel_launch(void* const* d_in, const int* in_sizes, int n_in,
                              void* d_out, int out_size, void* d_ws, size_t ws_size,
                              hipStream_t stream) {
  const float* inp = (const float*)d_in[0];
  const float* h   = (const float*)d_in[1];
  const float* c   = (const float*)d_in[2];
  const float* Wi  = (const float*)d_in[3];
  const float* Wf  = (const float*)d_in[4];
  const float* Wc  = (const float*)d_in[5];
  const float* Wo  = (const float*)d_in[6];
  const float* bi  = (const float*)d_in[7];
  const float* bf  = (const float*)d_in[8];
  const float* bc  = (const float*)d_in[9];
  const float* bo  = (const float*)d_in[10];

  // Workspace layout: [0, 32MB) lin f16 [T][B][D]; [32MB, 48MB) w16 [4][H][D]
  _Float16* lin = (_Float16*)d_ws;
  _Float16* w16 = (_Float16*)((char*)d_ws +
                    (size_t)TT * BB * DD * sizeof(_Float16));

  spikegen_kernel<<<(BB * DD) / 256, 256, 0, stream>>>(inp, h, lin);
  wconvert_kernel<<<(4 * HH * DD) / 256, 256, 0, stream>>>(Wi, Wf, Wc, Wo, w16);

  dim3 grid(HH / 64, BB / 16);   // (16, 16)
  fused_gemm_scan_kernel<<<grid, 128, 0, stream>>>(
      lin, w16, c, bi, bf, bc, bo, (float*)d_out);
}
